// HGTLayer_75737453298010
// MI455X (gfx1250) — matrix operand order repaired and checked by reading, not compiled, not run on hardware
//
#include <hip/hip_runtime.h>
#include <hip/hip_bf16.h>

// ---------------------------------------------------------------------------
// CDNA5 (gfx1250) HGT layer.
//  - fp32 -> bf16 pre-conversion pass for features/weights (done once; x_paper
//    feeds 4 GEMMs, W feeds every m-tile, so this removes all per-WMMA cvts)
//  - node-level projections via v_wmma_f32_16x16x32_bf16, A tile staged into
//    LDS in WMMA *fragment order* using GLOBAL_LOAD_ASYNC_TO_LDS_B128
//    (ASYNCcnt) so the inner loop is pure ds_load/global_load + WMMA
//  - edge phase: gather -> dot -> sigmoid -> scaled f32 atomic scatter
//    (feature tables are L2-resident: 192MB L2 holds all of K/Q/V)
//  - output projection (WMMA) + wave-per-row layernorm
//  - authors receive no edges -> y_author = LN(bO) broadcast
// ---------------------------------------------------------------------------

typedef __attribute__((ext_vector_type(16))) __bf16 v16bf;
typedef __attribute__((ext_vector_type(8)))  __bf16 v8bf;
typedef __attribute__((ext_vector_type(8)))  float  v8f;

#define DIM 256

// ---------------------------------------------------------------------------
// fp32 -> bf16 conversion, 8 elements per thread-iteration (n % 8 == 0).
// ---------------------------------------------------------------------------
__global__ __launch_bounds__(256)
void cvt_f32_to_bf16_kernel(const float* __restrict__ src, __bf16* __restrict__ dst,
                            size_t n)
{
    const size_t stride = (size_t)gridDim.x * blockDim.x;
    for (size_t i = (size_t)blockIdx.x * blockDim.x + threadIdx.x; i * 8 < n; i += stride) {
        float4 a = ((const float4*)src)[2 * i];
        float4 b = ((const float4*)src)[2 * i + 1];
        v8bf o;
        o[0] = (__bf16)a.x; o[1] = (__bf16)a.y; o[2] = (__bf16)a.z; o[3] = (__bf16)a.w;
        o[4] = (__bf16)b.x; o[5] = (__bf16)b.y; o[6] = (__bf16)b.z; o[7] = (__bf16)b.w;
        ((v8bf*)dst)[i] = o;
    }
}

// ---------------------------------------------------------------------------
// C[M,256] = A[M,256] @ W[256,256]^T + bias   (A, W already bf16; fp32 accum)
//
// 128-thread block = 4 waves; grid = (M/16, 4); wave w owns n-tile
// blockIdx.y*4+w. The 16x256 A tile is staged into LDS pre-swizzled into the
// contiguous-K A-fragment layout (same k order as the B fragment):
//   lane L (fragment row = L&15, half = L>>4, kbA = 16*half):
//     element i<8  -> K = kt*32 + kbA + i
//     element i>=8 -> K = kt*32 + kbA + 8 + (i-8)
// Every K-aligned 8-element group of a row maps to one contiguous 16B chunk of
// the fragment layout, so staging = per-lane 16B async copies (scattered LDS
// addresses are native to GLOBAL_LOAD_ASYNC_TO_LDS) and the inner-loop A read
// is a single aligned 32B LDS load.
// ---------------------------------------------------------------------------
__global__ __launch_bounds__(128)
void gemm256_bias_wmma(const __bf16* __restrict__ A, const __bf16* __restrict__ W,
                       const float* __restrict__ bias, float* __restrict__ C, int M)
{
    __shared__ __bf16 Afrag[8 * 32 * 16];   // [kt][lane][16], 8KB, fragment order

    const int tid = threadIdx.x;
    const int m0  = blockIdx.x * 16;
    if (m0 >= M) return;

    // ---- async-stage A tile (16 rows x 256) into fragment-order LDS ----
    #pragma unroll
    for (int it = 0; it < 4; ++it) {
        const int ch  = tid + it * 128;       // chunk 0..511
        const int row = ch >> 5;              // 32 chunks per row
        const int g   = ch & 31;              // aligned 8-elem K-group
        const int kt  = g >> 2;
        const int h   = (g >> 1) & 1;         // target half-wave
        const int ib  = (g & 1) * 8;          // element base within fragment
        const int ln  = row + 16 * h;         // target lane
        const unsigned lds = (unsigned)(size_t)&Afrag[(kt * 32 + ln) * 16 + ib];
        const unsigned long long ga =
            (unsigned long long)(size_t)(A + (size_t)(m0 + row) * DIM + g * 8);
        asm volatile("global_load_async_to_lds_b128 %0, %1, off"
                     :: "v"(lds), "v"(ga) : "memory");
    }
    asm volatile("s_wait_asynccnt 0" ::: "memory");
    __syncthreads();

    const int lane = tid & 31;
    const int wave = tid >> 5;
    const int n0   = (blockIdx.y * 4 + wave) * 16;
    const int col  = lane & 15;               // N index within tile
    const int half = lane >> 4;               // 0: lanes 0-15, 1: lanes 16-31
    const int kbB  = half * 16;               // B-frag contiguous K base

    // Bias depends only on the column -> same value in all 8 accum elements.
    const float bv = bias[n0 + col];
    v8f c;
    #pragma unroll
    for (int i = 0; i < 8; ++i) c[i] = bv;

    // B = W^T: column (n0+col) of B is row (n0+col) of W, K contiguous bf16.
    const __bf16* wrow = W + (size_t)(n0 + col) * DIM + kbB;

    #pragma unroll
    for (int kt = 0; kt < 8; ++kt) {
        v16bf a = *(const v16bf*)&Afrag[(kt * 32 + lane) * 16];
        v16bf b = *(const v16bf*)(wrow + kt * 32);
        c = __builtin_amdgcn_wmma_f32_16x16x32_bf16(
                /*neg_a=*/false, a, /*neg_b=*/false, b,
                /*c_mod=*/(short)0, c, /*reuse_a=*/false, /*reuse_b=*/false);
    }

    // C/D layout: element v -> row (v + 8*half), col = lane&15.
    float* crow = C + (size_t)m0 * DIM + n0 + col;
    #pragma unroll
    for (int v = 0; v < 8; ++v) {
        crow[(size_t)(v + half * 8) * DIM] = c[v];
    }
}

// ---------------------------------------------------------------------------
// Edge phase: one wave per edge.
//   attn = sigmoid( dot(Q[dst], K[src]) / sqrt(32) );  agg[dst] += attn * V[src]
// ---------------------------------------------------------------------------
__global__ __launch_bounds__(256)
void edge_msg_kernel(const int* __restrict__ src, const int* __restrict__ dst,
                     const float* __restrict__ K, const float* __restrict__ Q,
                     const float* __restrict__ V, float* __restrict__ agg, int E)
{
    const int lane = threadIdx.x & 31;
    const int e    = blockIdx.x * 8 + (threadIdx.x >> 5);
    if (e >= E) return;

    const int s = src[e];
    const int d = dst[e];

    const float4* k4 = (const float4*)(K + (size_t)s * DIM);
    const float4* q4 = (const float4*)(Q + (size_t)d * DIM);
    float4 ka = k4[lane], kb = k4[lane + 32];
    float4 qa = q4[lane], qb = q4[lane + 32];

    float p = ka.x*qa.x + ka.y*qa.y + ka.z*qa.z + ka.w*qa.w
            + kb.x*qb.x + kb.y*qb.y + kb.z*qb.z + kb.w*qb.w;
    #pragma unroll
    for (int off = 16; off > 0; off >>= 1) p += __shfl_xor(p, off, 32);

    const float attn = 1.0f / (1.0f + __expf(-p * 0.17677669529663687f)); // 1/sqrt(32)

    const float4* v4 = (const float4*)(V + (size_t)s * DIM);
    float4 va = v4[lane], vb = v4[lane + 32];

    float* out = agg + (size_t)d * DIM;
    const int c0 = 4 * lane;
    atomicAdd(&out[c0 + 0],       attn * va.x);
    atomicAdd(&out[c0 + 1],       attn * va.y);
    atomicAdd(&out[c0 + 2],       attn * va.z);
    atomicAdd(&out[c0 + 3],       attn * va.w);
    atomicAdd(&out[128 + c0 + 0], attn * vb.x);
    atomicAdd(&out[128 + c0 + 1], attn * vb.y);
    atomicAdd(&out[128 + c0 + 2], attn * vb.z);
    atomicAdd(&out[128 + c0 + 3], attn * vb.w);
}

// ---------------------------------------------------------------------------
// LayerNorm over rows of H[nrows,256]; one wave per row.
// ---------------------------------------------------------------------------
__global__ __launch_bounds__(256)
void layernorm_rows_kernel(const float* __restrict__ H, const float* __restrict__ gamma,
                           const float* __restrict__ beta, float* __restrict__ Y, int nrows)
{
    const int lane = threadIdx.x & 31;
    const int row  = blockIdx.x * 8 + (threadIdx.x >> 5);
    if (row >= nrows) return;

    const float4* h4 = (const float4*)(H + (size_t)row * DIM);
    float4 a = h4[lane], b = h4[lane + 32];

    float s  = a.x + a.y + a.z + a.w + b.x + b.y + b.z + b.w;
    float ss = a.x*a.x + a.y*a.y + a.z*a.z + a.w*a.w
             + b.x*b.x + b.y*b.y + b.z*b.z + b.w*b.w;
    #pragma unroll
    for (int off = 16; off > 0; off >>= 1) {
        s  += __shfl_xor(s,  off, 32);
        ss += __shfl_xor(ss, off, 32);
    }
    const float mu  = s  * (1.0f / 256.0f);
    const float var = ss * (1.0f / 256.0f) - mu * mu;
    const float rs  = rsqrtf(var + 1e-5f);

    const float4* g4  = (const float4*)gamma;
    const float4* be4 = (const float4*)beta;
    float4* y4 = (float4*)(Y + (size_t)row * DIM);

    float4 g = g4[lane], be = be4[lane];
    float4 o;
    o.x = (a.x - mu) * rs * g.x + be.x;
    o.y = (a.y - mu) * rs * g.y + be.y;
    o.z = (a.z - mu) * rs * g.z + be.z;
    o.w = (a.w - mu) * rs * g.w + be.w;
    y4[lane] = o;

    g = g4[lane + 32]; be = be4[lane + 32];
    o.x = (b.x - mu) * rs * g.x + be.x;
    o.y = (b.y - mu) * rs * g.y + be.y;
    o.z = (b.z - mu) * rs * g.z + be.z;
    o.w = (b.w - mu) * rs * g.w + be.w;
    y4[lane + 32] = o;
}

// ---------------------------------------------------------------------------
// Authors receive no edges -> agg_author == 0 -> y_author rows are all LN(bO).
// ---------------------------------------------------------------------------
__global__ __launch_bounds__(256)
void author_out_kernel(const float* __restrict__ bO, const float* __restrict__ gamma,
                       const float* __restrict__ beta, float* __restrict__ Y, int nrows)
{
    __shared__ float rsum[256];
    __shared__ float rssq[256];
    const int t = threadIdx.x;
    const float x = bO[t];
    rsum[t] = x; rssq[t] = x * x;
    __syncthreads();
    #pragma unroll
    for (int s = 128; s > 0; s >>= 1) {
        if (t < s) { rsum[t] += rsum[t + s]; rssq[t] += rssq[t + s]; }
        __syncthreads();
    }
    const float mu  = rsum[0] * (1.0f / 256.0f);
    const float var = rssq[0] * (1.0f / 256.0f) - mu * mu;
    const float y   = (x - mu) * rsqrtf(var + 1e-5f) * gamma[t] + beta[t];

    for (int row = blockIdx.x; row < nrows; row += gridDim.x)
        Y[(size_t)row * DIM + t] = y;
}

__global__ __launch_bounds__(256)
void zero_f4_kernel(float4* __restrict__ p, size_t n4)
{
    const size_t stride = (size_t)gridDim.x * blockDim.x;
    for (size_t i = (size_t)blockIdx.x * blockDim.x + threadIdx.x; i < n4; i += stride)
        p[i] = make_float4(0.f, 0.f, 0.f, 0.f);
}

// ---------------------------------------------------------------------------
extern "C" void kernel_launch(void* const* d_in, const int* in_sizes, int n_in,
                              void* d_out, int out_size, void* d_ws, size_t ws_size,
                              hipStream_t stream)
{
    const float* x_paper  = (const float*)d_in[0];
    const float* x_author = (const float*)d_in[1];
    const int*   src_c    = (const int*)d_in[2];
    const int*   dst_c    = (const int*)d_in[3];
    const int*   src_w    = (const int*)d_in[4];
    const int*   dst_w    = (const int*)d_in[5];
    const float* WK1 = (const float*)d_in[6];   const float* bK1 = (const float*)d_in[7];
    const float* WQ1 = (const float*)d_in[8];   const float* bQ1 = (const float*)d_in[9];
    const float* WV1 = (const float*)d_in[10];  const float* bV1 = (const float*)d_in[11];
    const float* WK2 = (const float*)d_in[12];  const float* bK2 = (const float*)d_in[13];
    const float* WQ2 = (const float*)d_in[14];  const float* bQ2 = (const float*)d_in[15];
    const float* WV2 = (const float*)d_in[16];  const float* bV2 = (const float*)d_in[17];
    const float* WO  = (const float*)d_in[18];  const float* bO  = (const float*)d_in[19];
    const float* gamma = (const float*)d_in[20];
    const float* beta  = (const float*)d_in[21];

    const int NP = in_sizes[0] / DIM;   // 100000
    const int NA = in_sizes[1] / DIM;   // 50000
    const int E1 = in_sizes[2];
    const int E2 = in_sizes[4];

    const size_t np = (size_t)NP * DIM;
    const size_t na = (size_t)NA * DIM;
    const size_t wsz = (size_t)DIM * DIM;  // one weight matrix

    // ---- fp32 workspace ----
    float* ws  = (float*)d_ws;
    float* Kp  = ws;            // [NP,256]
    float* Qp  = Kp  + np;      // [NP,256]
    float* Vp  = Qp  + np;      // [NP,256]
    float* Qp2 = Vp  + np;      // [NP,256]
    float* Ka  = Qp2 + np;      // [NA,256]
    float* Va  = Ka  + na;      // [NA,256]
    float* agg = Va  + na;      // [NP,256]
    float* h   = Kp;            // reuse Kp for the output-projection result

    // ---- bf16 workspace (after fp32 region) ----
    __bf16* Xp_bf  = (__bf16*)(agg + np);   // [NP,256]
    __bf16* Xa_bf  = Xp_bf + np;            // [NA,256]
    __bf16* Agg_bf = Xa_bf + na;            // [NP,256]
    __bf16* Wbf    = Agg_bf + np;           // 7 x [256,256]
    __bf16* WK1b = Wbf + 0 * wsz; __bf16* WQ1b = Wbf + 1 * wsz;
    __bf16* WV1b = Wbf + 2 * wsz; __bf16* WK2b = Wbf + 3 * wsz;
    __bf16* WQ2b = Wbf + 4 * wsz; __bf16* WV2b = Wbf + 5 * wsz;
    __bf16* WOb  = Wbf + 6 * wsz;

    // 1) zero the aggregation buffer (ws is poisoned by the harness)
    zero_f4_kernel<<<4096, 256, 0, stream>>>((float4*)agg, np / 4);

    // 2) one-time fp32 -> bf16 conversion of features and weights
    cvt_f32_to_bf16_kernel<<<4096, 256, 0, stream>>>(x_paper,  Xp_bf, np);
    cvt_f32_to_bf16_kernel<<<2048, 256, 0, stream>>>(x_author, Xa_bf, na);
    cvt_f32_to_bf16_kernel<<<64, 256, 0, stream>>>(WK1, WK1b, wsz);
    cvt_f32_to_bf16_kernel<<<64, 256, 0, stream>>>(WQ1, WQ1b, wsz);
    cvt_f32_to_bf16_kernel<<<64, 256, 0, stream>>>(WV1, WV1b, wsz);
    cvt_f32_to_bf16_kernel<<<64, 256, 0, stream>>>(WK2, WK2b, wsz);
    cvt_f32_to_bf16_kernel<<<64, 256, 0, stream>>>(WQ2, WQ2b, wsz);
    cvt_f32_to_bf16_kernel<<<64, 256, 0, stream>>>(WV2, WV2b, wsz);
    cvt_f32_to_bf16_kernel<<<64, 256, 0, stream>>>(WO,  WOb,  wsz);

    // 3) node-level projections (bf16 WMMA GEMMs, async-staged A tiles)
    const dim3 gP((NP + 15) / 16, 4), gA((NA + 15) / 16, 4);
    gemm256_bias_wmma<<<gP, 128, 0, stream>>>(Xp_bf, WK1b, bK1, Kp,  NP);
    gemm256_bias_wmma<<<gP, 128, 0, stream>>>(Xp_bf, WQ1b, bQ1, Qp,  NP);
    gemm256_bias_wmma<<<gP, 128, 0, stream>>>(Xp_bf, WV1b, bV1, Vp,  NP);
    gemm256_bias_wmma<<<gP, 128, 0, stream>>>(Xp_bf, WQ2b, bQ2, Qp2, NP);
    gemm256_bias_wmma<<<gA, 128, 0, stream>>>(Xa_bf, WK2b, bK2, Ka,  NA);
    gemm256_bias_wmma<<<gA, 128, 0, stream>>>(Xa_bf, WV2b, bV2, Va,  NA);

    // 4) edge phase: gather -> dot -> sigmoid -> scaled atomic scatter
    edge_msg_kernel<<<(E1 + 7) / 8, 256, 0, stream>>>(src_c, dst_c, Kp, Qp,  Vp, agg, E1);
    edge_msg_kernel<<<(E2 + 7) / 8, 256, 0, stream>>>(src_w, dst_w, Ka, Qp2, Va, agg, E2);

    // 5) output projection (WMMA) + layernorm
    cvt_f32_to_bf16_kernel<<<4096, 256, 0, stream>>>(agg, Agg_bf, np);
    gemm256_bias_wmma<<<gP, 128, 0, stream>>>(Agg_bf, WOb, bO, h, NP);
    layernorm_rows_kernel<<<(NP + 7) / 8, 256, 0, stream>>>(h, gamma, beta, (float*)d_out, NP);

    // 6) authors: agg is zero -> broadcast LN(bO)
    author_out_kernel<<<2048, 256, 0, stream>>>(bO, gamma, beta,
                                                (float*)d_out + np, NA);
}
